// PointNetPPEncoderDecoder_65953517797452
// MI455X (gfx1250) — compile-verified, matrix operand order checked by reading, not run on hardware
//
#include <hip/hip_runtime.h>

typedef __attribute__((ext_vector_type(16))) _Float16 v16h;
typedef __attribute__((ext_vector_type(8)))  _Float16 v8h;
typedef __attribute__((ext_vector_type(8)))  float    v8f;

#define BN_SCALE 0.99999500003749974f /* 1/sqrt(1+1e-5) */

// ---------------------------------------------------------------------------
// Split input (B,N,9) -> xyz0 [B][N][3] and pts0 [B][9][N]
// ---------------------------------------------------------------------------
__global__ void k_split_input(const float* __restrict__ x, float* __restrict__ xyz0,
                              float* __restrict__ pts0, int N, int total) {
  int t = blockIdx.x * blockDim.x + threadIdx.x;
  if (t >= total) return;
  int b = t / N, n = t % N;
  const float* src = x + ((size_t)b * N + n) * 9;
  float* xo = xyz0 + ((size_t)b * N + n) * 3;
  float* po = pts0 + (size_t)b * 9 * N;
#pragma unroll
  for (int c = 0; c < 3; ++c) xo[c] = src[c];
#pragma unroll
  for (int c = 0; c < 9; ++c) po[(size_t)c * N + n] = src[c];
}

// ---------------------------------------------------------------------------
// Farthest point sampling: one block per batch (mirrors reference scan order).
// ---------------------------------------------------------------------------
__global__ void k_fps(const float* __restrict__ xyz, int* __restrict__ fps_idx,
                      int N, int J) {
  __shared__ float s_dist[4096];
  __shared__ float s_val[256];
  __shared__ int   s_idx[256];
  int b = blockIdx.x, tid = threadIdx.x;
  const float* xb = xyz + (size_t)b * N * 3;
  for (int i = tid; i < N; i += blockDim.x) s_dist[i] = 1e10f;
  __syncthreads();
  int far = 0;
  for (int it = 0; it < J; ++it) {
    if (tid == 0) fps_idx[b * J + it] = far;
    float cx = xb[far * 3 + 0], cy = xb[far * 3 + 1], cz = xb[far * 3 + 2];
    float best = -1.0f; int bi = 0;
    for (int i = tid; i < N; i += blockDim.x) {
      float dx = xb[i * 3 + 0] - cx;
      float dy = xb[i * 3 + 1] - cy;
      float dz = xb[i * 3 + 2] - cz;
      float d = dx * dx + dy * dy + dz * dz;
      float nd = fminf(s_dist[i], d);
      s_dist[i] = nd;
      if (nd > best) { best = nd; bi = i; }
    }
    s_val[tid] = best; s_idx[tid] = bi;
    __syncthreads();
    for (int off = blockDim.x / 2; off > 0; off >>= 1) {
      if (tid < off) {
        float v2 = s_val[tid + off]; int i2 = s_idx[tid + off];
        if (v2 > s_val[tid] || (v2 == s_val[tid] && i2 < s_idx[tid])) {
          s_val[tid] = v2; s_idx[tid] = i2;
        }
      }
      __syncthreads();
    }
    far = s_idx[0];
    __syncthreads();
  }
}

__global__ void k_gather_xyz(const float* __restrict__ xyz, const int* __restrict__ fidx,
                             float* __restrict__ nxyz, int N, int J, int total) {
  int t = blockIdx.x * blockDim.x + threadIdx.x;
  if (t >= total) return;
  int b = t / J, j = t % J;
  int s = fidx[b * J + j];
#pragma unroll
  for (int c = 0; c < 3; ++c)
    nxyz[((size_t)b * J + j) * 3 + c] = xyz[((size_t)b * N + s) * 3 + c];
}

// Ball query: first S points (ascending index) within radius; pad with first.
__global__ void k_ball(const float* __restrict__ xyz, const float* __restrict__ nxyz,
                       int* __restrict__ idx, int N, int J, int S, float r2, int total) {
  int t = blockIdx.x * blockDim.x + threadIdx.x;
  if (t >= total) return;
  int b = t / J, j = t % J;
  const float* xb = xyz + (size_t)b * N * 3;
  float cx = nxyz[((size_t)b * J + j) * 3 + 0];
  float cy = nxyz[((size_t)b * J + j) * 3 + 1];
  float cz = nxyz[((size_t)b * J + j) * 3 + 2];
  int* out = idx + ((size_t)b * J + j) * S;
  int cnt = 0;
  for (int n = 0; n < N && cnt < S; ++n) {
    float dx = xb[n * 3 + 0] - cx;
    float dy = xb[n * 3 + 1] - cy;
    float dz = xb[n * 3 + 2] - cz;
    float d2 = dx * dx + dy * dy + dz * dz;
    if (d2 <= r2) out[cnt++] = n;
  }
  int f = (cnt > 0) ? out[0] : 0;
  for (; cnt < S; ++cnt) out[cnt] = f;
}

// Grouped activation [B][Cin_pad][S*J], pos = k*J + j; pad channels zeroed.
__global__ void k_group(const float* __restrict__ xyz, const float* __restrict__ nxyz,
                        const float* __restrict__ pts, const int* __restrict__ idx,
                        float* __restrict__ act, int N, int J, int S, int Cpts,
                        int Cin_pad, int total) {
  int t = blockIdx.x * blockDim.x + threadIdx.x;
  if (t >= total) return;
  int b = t / (S * J);
  int r = t % (S * J);
  int k = r / J;
  int j = r % J;
  int s = idx[((size_t)b * J + j) * S + k];
  int P = S * J;
  int pos = k * J + j;
  float* ab = act + (size_t)b * Cin_pad * P;
#pragma unroll
  for (int c = 0; c < 3; ++c)
    ab[(size_t)c * P + pos] =
        xyz[((size_t)b * N + s) * 3 + c] - nxyz[((size_t)b * J + j) * 3 + c];
  const float* pb = pts + (size_t)b * Cpts * N;
  for (int c = 0; c < Cpts; ++c)
    ab[(size_t)(3 + c) * P + pos] = pb[(size_t)c * N + s];
  for (int c = Cpts + 3; c < Cin_pad; ++c)
    ab[(size_t)c * P + pos] = 0.0f;
}

// ---------------------------------------------------------------------------
// Pack weights to f16 in the CDNA5 16-bit A-operand lane layout:
//   out[((ot*Ktiles + kt)*32 + lane)*16 + e],
//   m = lane&15, kb = (lane>>4)*8, k = kt*32 + kb + e + (e>=8 ? 8 : 0)
// Zero-padded for k >= Cin.
// ---------------------------------------------------------------------------
__global__ void k_pack_w(const float* __restrict__ W, _Float16* __restrict__ out,
                         int Cin, int Ktiles, int total) {
  int t = blockIdx.x * blockDim.x + threadIdx.x;
  if (t >= total) return;
  int lane = t & 31;
  int gidx = t >> 5;
  int kt = gidx % Ktiles;
  int ot = gidx / Ktiles;
  int m  = lane & 15;
  int kb = (lane >> 4) * 8;
  _Float16* dst = out + (size_t)t * 16;
#pragma unroll
  for (int e = 0; e < 16; ++e) {
    int k = kt * 32 + kb + e + ((e >= 8) ? 8 : 0);
    float v = (k < Cin) ? W[(size_t)(ot * 16 + m) * Cin + k] : 0.0f;
    dst[e] = (_Float16)v;
  }
}

// ---------------------------------------------------------------------------
// Fused pointwise-MLP via WMMA. Block = 128 threads (4 waves) -> 32x32 output
// tile; wave w: mrow=w>>1, ncol=w&1. K tiled by 32, no edge cases (all padded).
// X panel staged through LDS with b128 global loads; A is pre-packed f16.
//   Y[o,p] = relu((sum_c W[o,c]X[c,p] + bias[o]) * g[o]*BN_SCALE + beta[o])
// ---------------------------------------------------------------------------
__global__ __launch_bounds__(128) void k_mlp_wmma(
    const _Float16* __restrict__ Apk, const float* __restrict__ X,
    const float* __restrict__ bias, const float* __restrict__ g,
    const float* __restrict__ beta, float* __restrict__ Y,
    int Cin_pad, int Cout, int P) {
  __shared__ _Float16 sB[32][72];   // [n][k], row stride 144B (16B aligned)
  int tid  = threadIdx.x;
  int lane = tid & 31;
  int wv   = tid >> 5;
  int mrow = wv >> 1;
  int ncol = wv & 1;
  int p0 = blockIdx.x * 32;
  int o0 = blockIdx.y * 32;
  int b  = blockIdx.z;
  int Ktiles = Cin_pad >> 5;
  const float* Xb = X + (size_t)b * Cin_pad * P;
  float* Yb = Y + (size_t)b * Cout * P;
  int lm   = lane & 15;
  int half = lane >> 4;
  int kb2  = half * 8;
  int kld = tid & 31;           // staging: k row
  int nch = (tid >> 5) * 8;     // staging: n chunk (8 cols)
  const _Float16* Abase =
      Apk + ((size_t)(blockIdx.y * 2 + mrow) * Ktiles * 32 + lane) * 16;

  v8f acc = {};
  for (int kt = 0; kt < Ktiles; ++kt) {
    const float* xr = Xb + (size_t)(kt * 32 + kld) * P + p0 + nch;
    float4 x0 = *(const float4*)xr;
    float4 x1 = *(const float4*)(xr + 4);
    sB[nch + 0][kld] = (_Float16)x0.x;
    sB[nch + 1][kld] = (_Float16)x0.y;
    sB[nch + 2][kld] = (_Float16)x0.z;
    sB[nch + 3][kld] = (_Float16)x0.w;
    sB[nch + 4][kld] = (_Float16)x1.x;
    sB[nch + 5][kld] = (_Float16)x1.y;
    sB[nch + 6][kld] = (_Float16)x1.z;
    sB[nch + 7][kld] = (_Float16)x1.w;
    __syncthreads();
    v16h av = *(const v16h*)(Abase + (size_t)kt * 512);
    const _Float16* brow = &sB[ncol * 16 + lm][0];
    v8h blo = *(const v8h*)(brow + kb2);
    v8h bhi = *(const v8h*)(brow + kb2 + 16);
    v16h bv;
#pragma unroll
    for (int e = 0; e < 8; ++e) { bv[e] = blo[e]; bv[e + 8] = bhi[e]; }
    acc = __builtin_amdgcn_wmma_f32_16x16x32_f16(
        false, av, false, bv, (short)0, acc, false, false);
    __syncthreads();
  }
#pragma unroll
  for (int r = 0; r < 8; ++r) {
    int o = o0 + mrow * 16 + half * 8 + r;
    float v = acc[r] + bias[o];
    v = v * (g[o] * BN_SCALE) + beta[o];
    v = fmaxf(v, 0.0f);
    Yb[(size_t)o * P + p0 + ncol * 16 + lm] = v;
  }
}

// Max over nsample: in [B][C][S*J] (pos=k*J+j) -> out [B][C][J]
__global__ void k_maxpool(const float* __restrict__ y, float* __restrict__ out,
                          int C, int J, int S, int total) {
  int t = blockIdx.x * blockDim.x + threadIdx.x;
  if (t >= total) return;
  int b = t / (C * J);
  int r = t % (C * J);
  int c = r / J;
  int j = r % J;
  int P = S * J;
  const float* yb = y + (size_t)b * C * P + (size_t)c * P;
  float m = -3.4e38f;
  for (int k = 0; k < S; ++k) m = fmaxf(m, yb[(size_t)k * J + j]);
  out[(size_t)b * C * J + (size_t)c * J + j] = m;
}

// 3-NN inverse-distance interpolation -> channels [Cofs, Cofs+C2) of concat.
__global__ void k_interp(const float* __restrict__ xyz1, const float* __restrict__ xyz2,
                         const float* __restrict__ p2, float* __restrict__ out,
                         int N1, int N2, int C2, int Cofs, int Ctot, int total) {
  int t = blockIdx.x * blockDim.x + threadIdx.x;
  if (t >= total) return;
  int b = t / N1, n = t % N1;
  const float* a = xyz1 + ((size_t)b * N1 + n) * 3;
  float ax = a[0], ay = a[1], az = a[2];
  float d0 = 1e30f, d1 = 1e30f, d2 = 1e30f;
  int i0 = 0, i1 = 0, i2 = 0;
  const float* x2 = xyz2 + (size_t)b * N2 * 3;
  for (int m = 0; m < N2; ++m) {
    float dx = ax - x2[m * 3 + 0];
    float dy = ay - x2[m * 3 + 1];
    float dz = az - x2[m * 3 + 2];
    float d = dx * dx + dy * dy + dz * dz;
    if (d < d0)      { d2 = d1; i2 = i1; d1 = d0; i1 = i0; d0 = d; i0 = m; }
    else if (d < d1) { d2 = d1; i2 = i1; d1 = d;  i1 = m; }
    else if (d < d2) { d2 = d;  i2 = m; }
  }
  float w0 = 1.0f / (d0 + 1e-8f);
  float w1 = 1.0f / (d1 + 1e-8f);
  float w2 = 1.0f / (d2 + 1e-8f);
  float ws = w0 + w1 + w2;
  w0 /= ws; w1 /= ws; w2 /= ws;
  const float* pb = p2 + (size_t)b * C2 * N2;
  float* ob = out + (size_t)b * Ctot * N1;
  for (int c = 0; c < C2; ++c) {
    const float* row = pb + (size_t)c * N2;
    ob[(size_t)(Cofs + c) * N1 + n] = w0 * row[i0] + w1 * row[i1] + w2 * row[i2];
  }
}

__global__ void k_copy_ch(const float* __restrict__ src, float* __restrict__ dst,
                          int C1, int N1, int Ctot, int total) {
  int t = blockIdx.x * blockDim.x + threadIdx.x;
  if (t >= total) return;
  int b = t / (C1 * N1);
  int r = t % (C1 * N1);
  int c = r / N1;
  int n = r % N1;
  dst[(size_t)b * Ctot * N1 + (size_t)c * N1 + n] =
      src[(size_t)b * C1 * N1 + (size_t)c * N1 + n];
}

// Width-3 conv + BN + sigmoid + coef gate + relu + sum -> out[b].
__global__ void k_final(const float* __restrict__ h, const float* __restrict__ W2,
                        const float* __restrict__ b2, const float* __restrict__ g2,
                        const float* __restrict__ be2, const float* __restrict__ coef,
                        float* __restrict__ out, int N) {
  __shared__ float sw[384];
  __shared__ float sred[256];
  int b = blockIdx.x, tid = threadIdx.x;
  const float* hb = h + (size_t)b * 128 * N;
  for (int i = tid; i < 384; i += blockDim.x) sw[i] = W2[i];
  __syncthreads();
  float bb = b2[0];
  float gg = g2[0] * BN_SCALE;
  float be = be2[0];
  float partial = 0.0f;
  for (int n = tid; n < N; n += blockDim.x) {
    float acc = bb;
    for (int c = 0; c < 128; ++c) {
      const float* row = hb + (size_t)c * N;
      float xm = (n > 0)     ? row[n - 1] : 0.0f;
      float x0 = row[n];
      float xp = (n < N - 1) ? row[n + 1] : 0.0f;
      acc += sw[c * 3 + 0] * xm + sw[c * 3 + 1] * x0 + sw[c * 3 + 2] * xp;
    }
    acc = acc * gg + be;
    acc = 1.0f / (1.0f + __expf(-acc));
    acc = fmaxf(acc * coef[(size_t)b * N + n], 0.0f);
    partial += acc;
  }
  sred[tid] = partial;
  __syncthreads();
  for (int off = 128; off > 0; off >>= 1) {
    if (tid < off) sred[tid] += sred[tid + off];
    __syncthreads();
  }
  if (tid == 0) out[b] = sred[0];
}

// ---------------------------------------------------------------------------
// Host orchestration
// ---------------------------------------------------------------------------
extern "C" void kernel_launch(void* const* d_in, const int* in_sizes, int n_in,
                              void* d_out, int out_size, void* d_ws, size_t ws_size,
                              hipStream_t stream) {
  (void)in_sizes; (void)n_in; (void)out_size; (void)ws_size;
  const int B = 16, N0 = 4096, S = 32;

  const float* xin  = (const float*)d_in[0];
  const float* coef = (const float*)d_in[1];

  const float *Wp[23], *bp[23], *gp[23], *betap[23];
  for (int i = 0; i < 23; ++i) {
    Wp[i]    = (const float*)d_in[2 + i * 4 + 0];
    bp[i]    = (const float*)d_in[2 + i * 4 + 1];
    gp[i]    = (const float*)d_in[2 + i * 4 + 2];
    betap[i] = (const float*)d_in[2 + i * 4 + 3];
  }

  // Layer table (22 GEMM layers: sa1..sa4, fp4..fp1, conv1)
  static const int CinA[22]  = {12, 32, 32,  67, 64, 64,  131, 128, 128,
                                259, 256, 256, 768, 256, 384, 256, 320, 256,
                                128, 128, 128, 128};
  static const int CoutA[22] = {32, 32, 64,  64, 64, 128, 128, 128, 256,
                                256, 256, 512, 256, 256, 256, 256, 256, 128,
                                128, 128, 128, 128};

  char* ws = (char*)d_ws;
  size_t cur = 0;
  auto allocB = [&](size_t bytes) -> void* {
    size_t a = (cur + 255) & ~(size_t)255;
    void* p = ws + a;
    cur = a + bytes;
    return p;
  };
  auto allocF = [&](size_t elems) -> float* { return (float*)allocB(elems * 4); };

  float* xyz0 = allocF((size_t)B * N0 * 3);
  float* pts0 = allocF((size_t)B * 9 * N0);
  float* xyz1 = allocF((size_t)B * 1024 * 3);
  float* xyz2 = allocF((size_t)B * 256 * 3);
  float* xyz3 = allocF((size_t)B * 64 * 3);
  float* xyz4 = allocF((size_t)B * 16 * 3);
  float* p1   = allocF((size_t)B * 64 * 1024);
  float* p2   = allocF((size_t)B * 128 * 256);
  float* p3   = allocF((size_t)B * 256 * 64);
  float* p4   = allocF((size_t)B * 512 * 16);
  float* p3n  = allocF((size_t)B * 256 * 64);
  float* p2n  = allocF((size_t)B * 256 * 256);
  float* p1n  = allocF((size_t)B * 128 * 1024);
  float* p0n  = allocF((size_t)B * 128 * 4096);
  int*   fidx = (int*)allocB((size_t)B * 1024 * 4);
  int*   bidx = (int*)allocB((size_t)B * 1024 * S * 4);
  float* bufA = allocF((size_t)B * 32 * 32768);   // grouped inputs / even layers
  float* bufB = allocF((size_t)B * 64 * 32768);   // odd layers (sa1 worst case)

  // Pre-pack all GEMM weights to f16 A-operand layout.
  _Float16* Apk[22];
  for (int li = 0; li < 22; ++li) {
    int cp = (CinA[li] + 31) & ~31;
    int Ktiles = cp >> 5;
    Apk[li] = (_Float16*)allocB((size_t)CoutA[li] * cp * 2);
    int tot = (CoutA[li] / 16) * Ktiles * 32;
    k_pack_w<<<dim3((tot + 255) / 256), dim3(256), 0, stream>>>(
        Wp[li], Apk[li], CinA[li], Ktiles, tot);
  }

  auto mlp = [&](const float* X, int li, float* Y, int P) {
    int Cout = CoutA[li];
    int cp = (CinA[li] + 31) & ~31;
    dim3 grid(P / 32, Cout / 32, B);
    k_mlp_wmma<<<grid, dim3(128), 0, stream>>>(
        Apk[li], X, bp[li], gp[li], betap[li], Y, cp, Cout, P);
  };

  auto sa = [&](const float* xyzIn, const float* ptsIn, int Nin, int Cpts, int J,
                float radius, int li0, float* xyzOut, float* ptsOut) {
    k_fps<<<dim3(B), dim3(256), 0, stream>>>(xyzIn, fidx, Nin, J);
    int BJ = B * J;
    k_gather_xyz<<<dim3((BJ + 255) / 256), dim3(256), 0, stream>>>(
        xyzIn, fidx, xyzOut, Nin, J, BJ);
    k_ball<<<dim3((BJ + 255) / 256), dim3(256), 0, stream>>>(
        xyzIn, xyzOut, bidx, Nin, J, S, radius * radius, BJ);
    int cinPad = (Cpts + 3 + 31) & ~31;
    int BJS = B * J * S;
    k_group<<<dim3((BJS + 255) / 256), dim3(256), 0, stream>>>(
        xyzIn, xyzOut, ptsIn, bidx, bufA, Nin, J, S, Cpts, cinPad, BJS);
    int P = S * J;
    mlp(bufA, li0 + 0, bufB, P);
    mlp(bufB, li0 + 1, bufA, P);
    mlp(bufA, li0 + 2, bufB, P);
    int c2 = CoutA[li0 + 2];
    int tot = B * c2 * J;
    k_maxpool<<<dim3((tot + 255) / 256), dim3(256), 0, stream>>>(
        bufB, ptsOut, c2, J, S, tot);
  };

  int totBN = B * N0;
  k_split_input<<<dim3((totBN + 255) / 256), dim3(256), 0, stream>>>(
      xin, xyz0, pts0, N0, totBN);

  sa(xyz0, pts0, N0,   9, 1024, 0.1f, 0, xyz1, p1);
  sa(xyz1, p1,  1024, 64,  256, 0.2f, 3, xyz2, p2);
  sa(xyz2, p2,   256, 128,  64, 0.4f, 6, xyz3, p3);
  sa(xyz3, p3,    64, 256,  16, 0.8f, 9, xyz4, p4);

  { // fp4: (l3, l4) -> p3n, cin 768 = 256 + 512
    int N1 = 64, N2 = 16, C1 = 256, C2 = 512, Ctot = 768;
    int BN1 = B * N1;
    k_interp<<<dim3((BN1 + 255) / 256), dim3(256), 0, stream>>>(
        xyz3, xyz4, p4, bufA, N1, N2, C2, C1, Ctot, BN1);
    int tot = B * C1 * N1;
    k_copy_ch<<<dim3((tot + 255) / 256), dim3(256), 0, stream>>>(
        p3, bufA, C1, N1, Ctot, tot);
    mlp(bufA, 12, bufB, N1);
    mlp(bufB, 13, p3n, N1);
  }
  { // fp3: (l2, l3') -> p2n, cin 384 = 128 + 256
    int N1 = 256, N2 = 64, C1 = 128, C2 = 256, Ctot = 384;
    int BN1 = B * N1;
    k_interp<<<dim3((BN1 + 255) / 256), dim3(256), 0, stream>>>(
        xyz2, xyz3, p3n, bufA, N1, N2, C2, C1, Ctot, BN1);
    int tot = B * C1 * N1;
    k_copy_ch<<<dim3((tot + 255) / 256), dim3(256), 0, stream>>>(
        p2, bufA, C1, N1, Ctot, tot);
    mlp(bufA, 14, bufB, N1);
    mlp(bufB, 15, p2n, N1);
  }
  { // fp2: (l1, l2') -> p1n, cin 320 = 64 + 256
    int N1 = 1024, N2 = 256, C1 = 64, C2 = 256, Ctot = 320;
    int BN1 = B * N1;
    k_interp<<<dim3((BN1 + 255) / 256), dim3(256), 0, stream>>>(
        xyz1, xyz2, p2n, bufA, N1, N2, C2, C1, Ctot, BN1);
    int tot = B * C1 * N1;
    k_copy_ch<<<dim3((tot + 255) / 256), dim3(256), 0, stream>>>(
        p1, bufA, C1, N1, Ctot, tot);
    mlp(bufA, 16, bufB, N1);
    mlp(bufB, 17, p1n, N1);
  }
  { // fp1: interp only (points1 = None), cin 128
    int N1 = 4096, N2 = 1024, C2 = 128, Ctot = 128;
    int BN1 = B * N1;
    k_interp<<<dim3((BN1 + 255) / 256), dim3(256), 0, stream>>>(
        xyz0, xyz1, p1n, bufA, N1, N2, C2, 0, Ctot, BN1);
    mlp(bufA, 18, bufB, N1);
    mlp(bufB, 19, bufA, N1);
    mlp(bufA, 20, p0n, N1);
  }

  // conv1 (pointwise 128x128 GEMM + BN + relu)
  mlp(p0n, 21, bufB, 4096);

  // conv2 (width-3) + BN + sigmoid + coef gate + relu + sum over points
  k_final<<<dim3(B), dim3(256), 0, stream>>>(
      bufB, Wp[22], bp[22], gp[22], betap[22], coef, (float*)d_out, N0);
}
